// SolvGNN_84499186581638
// MI455X (gfx1250) — matrix operand order repaired and checked by reading, not compiled, not run on hardware
//
#include <hip/hip_runtime.h>

// ---------------------------------------------------------------------------
// SolvGNN on MI455X (gfx1250): all dense contractions run through
// v_wmma_f32_16x16x32_bf16 (bf16 inputs, f32 accumulate). The 512MB per-edge
// weight tensor of NNConv is refactored into a single [8192 x 4224]x[4224 x 128]
// GEMM. GEMM blocks stage B slabs in LDS (double-buffered) so 4 waves share
// weight traffic and each wave computes a 16x64 tile (4 WMMAs per A fragment).
// Graph scatters use fp32 global atomics (L2-resident working set).
// ---------------------------------------------------------------------------

typedef __bf16 bf16_t;
typedef __attribute__((ext_vector_type(8)))  __bf16 bf16x8;
typedef __attribute__((ext_vector_type(16))) __bf16 bf16x16;
typedef __attribute__((ext_vector_type(8)))  float  floatx8;

enum : int {
  NN_   = 65536,     // nodes per side
  NE_   = 262144,    // directed edges per side
  HID_  = 128,
  BSZ_  = 2048,
  B2_   = 4096,      // 2B node-graph nodes
  E2_   = 8192,      // 4B meta-edges
  EH_   = 32,
  KIN_  = 74,
  KP1_  = 96,        // 74 padded to multiple of 32
  KPE_  = 4224       // 128*32 outer product + 128 bias lanes
};

// ---------------------------------------------------------------------------
// bf16 WMMA GEMM:  C[M,N] = A[M,K] * B[K,N] (+bias) (opt relu)
// A row-major bf16; Bt = B transposed ([N, K] row-major bf16).
// Block = 4 waves = 64x64 output tile. Per K-chunk (32):
//   - 64col x 32K B slab (4KB) staged cooperatively in LDS, double-buffered
//   - each wave: 1 A fragment (2x b128 global), 4 B fragments (ds_load), 4 WMMAs
// M,N multiples of 64; K multiple of 32. All control flow wave-uniform.
// ---------------------------------------------------------------------------
__global__ __launch_bounds__(128)
void wmma_gemm_bf16(const bf16_t* __restrict__ A, const bf16_t* __restrict__ Bt,
                    const float* __restrict__ bias, float* __restrict__ C,
                    int M, int N, int K, int fuse_relu)
{
  __shared__ bf16_t smem[2 * 64 * 32];        // double-buffered B slab (8KB)

  const int tid  = threadIdx.x;
  const int lane = tid & 31;
  const int wave = tid >> 5;
  const int gn   = N >> 6;                    // 64-col groups
  const int brow = blockIdx.x / gn;
  const int ngrp = blockIdx.x - brow * gn;
  const int tm   = brow * 4 + wave;           // 16-row tile per wave
  const int nbase = ngrp * 64;

  const int lhalf = lane >> 4;                // 0 | 1
  const int l15   = lane & 15;
  const long arow = (long)(tm * 16 + l15) * K;
  const int aoff  = lhalf * 8;                // A: K in {aoff..+7} U {16+aoff..}
  const int boff  = lhalf * 8;                // B: K in {boff..+7} U {16+boff..}

  // cooperative B staging: 2 x 16B segments per thread (LDS layout [col][32])
  const int col0 = (tid + 0)   >> 2, ko0 = ((tid + 0)   & 3) * 8;
  const int col1 = (tid + 128) >> 2, ko1 = ((tid + 128) & 3) * 8;
  const long bg0 = (long)(nbase + col0) * K + ko0;
  const long bg1 = (long)(nbase + col1) * K + ko1;

  // preload K-chunk 0
  *(bf16x8*)(smem + col0 * 32 + ko0) = *(const bf16x8*)(Bt + bg0);
  *(bf16x8*)(smem + col1 * 32 + ko1) = *(const bf16x8*)(Bt + bg1);
  __syncthreads();

  floatx8 acc[4] = {};
  int cur = 0;
  for (int kb = 0; kb < K; kb += 32) {
    const bool more = (kb + 32) < K;
    bf16x8 s0 = {}, s1 = {};
    if (more) {                                // stage next slab into registers
      s0 = *(const bf16x8*)(Bt + bg0 + kb + 32);
      s1 = *(const bf16x8*)(Bt + bg1 + kb + 32);
      __builtin_prefetch((const void*)(A + arow + kb + 32), 0, 3);
    }
    // A fragment (global)
    const bf16x8* ap = reinterpret_cast<const bf16x8*>(A + arow + kb + aoff);
    bf16x8 a0 = ap[0];
    bf16x8 a1 = ap[2];                         // +16 bf16 elements
    bf16x16 av;
#pragma unroll
    for (int j = 0; j < 8; ++j) { av[j] = a0[j]; av[j + 8] = a1[j]; }

    // 4 column tiles from LDS
    const bf16_t* sb = smem + cur * 2048;
#pragma unroll
    for (int ct = 0; ct < 4; ++ct) {
      const bf16x8* bp = reinterpret_cast<const bf16x8*>(sb + (ct * 16 + l15) * 32 + boff);
      bf16x8 b0 = bp[0];
      bf16x8 b1 = bp[2];
      bf16x16 bv;
#pragma unroll
      for (int j = 0; j < 8; ++j) { bv[j] = b0[j]; bv[j + 8] = b1[j]; }
      acc[ct] = __builtin_amdgcn_wmma_f32_16x16x32_bf16(
          /*neg_a=*/false, av, /*neg_b=*/false, bv,
          /*c_mod=*/(short)0, acc[ct], /*reuse_a=*/false, /*reuse_b=*/false);
    }
    if (more) {                                // commit next slab to other buffer
      bf16_t* sn = smem + (cur ^ 1) * 2048;
      *(bf16x8*)(sn + col0 * 32 + ko0) = s0;
      *(bf16x8*)(sn + col1 * 32 + ko1) = s1;
    }
    __syncthreads();
    cur ^= 1;
  }

  const int r0 = tm * 16 + lhalf * 8;
#pragma unroll
  for (int ct = 0; ct < 4; ++ct) {
    const int ncol = nbase + ct * 16 + l15;
    const float bs = bias ? bias[ncol] : 0.0f;
#pragma unroll
    for (int v = 0; v < 8; ++v) {
      float val = acc[ct][v] + bs;
      if (fuse_relu) val = fmaxf(val, 0.0f);
      C[(long)(r0 + v) * N + ncol] = val;
    }
  }
}

// ------------------------- elementwise / graph kernels ---------------------

__global__ void f32_to_bf16_pad(const float* __restrict__ src, bf16_t* __restrict__ dst,
                                int M, int Ks, int Kd, int relu)
{
  long i = (long)blockIdx.x * blockDim.x + threadIdx.x;
  if (i >= (long)M * Kd) return;
  int r = (int)(i / Kd), c = (int)(i - (long)r * Kd);
  float v = (c < Ks) ? src[(long)r * Ks + c] : 0.0f;
  if (relu) v = fmaxf(v, 0.0f);
  dst[i] = (bf16_t)v;
}

// W [K, Ncol] f32 row-major  ->  Wt [Ncol, Kpad] bf16 (zero-padded K)
__global__ void transpose_to_bf16(const float* __restrict__ W, bf16_t* __restrict__ Wt,
                                  int K, int Ncol, int Kpad)
{
  long i = (long)blockIdx.x * blockDim.x + threadIdx.x;
  if (i >= (long)Ncol * Kpad) return;
  int n = (int)(i / Kpad), k = (int)(i - (long)n * Kpad);
  float v = (k < K) ? W[(long)k * Ncol + n] : 0.0f;
  Wt[i] = (bf16_t)v;
}

// W2t[o, i*32+k] = en2_W[k, i*128+o];  W2t[o, 4096+i] = en2_b[i*128+o]
__global__ void build_w2t(const float* __restrict__ en2W, const float* __restrict__ en2b,
                          bf16_t* __restrict__ W2t)
{
  long i = (long)blockIdx.x * blockDim.x + threadIdx.x;
  if (i >= (long)HID_ * KPE_) return;
  int o = (int)(i / KPE_), kk = (int)(i - (long)o * KPE_);
  float v;
  if (kk < 4096) {
    int ii = kk >> 5, k = kk & 31;
    v = en2W[(long)k * 16384 + ii * 128 + o];
  } else {
    v = en2b[(kk - 4096) * 128 + o];
  }
  W2t[i] = (bf16_t)v;
}

__global__ void deg_init(float* deg, int n)
{
  int i = blockIdx.x * blockDim.x + threadIdx.x;
  if (i < n) deg[i] = 1.0f;
}

__global__ void deg_count(const int* __restrict__ dst, float* deg, int E)
{
  int i = blockIdx.x * blockDim.x + threadIdx.x;
  if (i < E) atomicAdd(&deg[dst[i]], 1.0f);
}

__global__ void deg_finalize(float* deg, int n)     // in-place deg -> deg^-1/2
{
  int i = blockIdx.x * blockDim.x + threadIdx.x;
  if (i < n) deg[i] = rsqrtf(deg[i]);
}

// out[n,f] = h[n,f]*dinv[n]^2 + bias[f]
__global__ void gcn_self(const float* __restrict__ h, const float* __restrict__ dinv,
                         const float* __restrict__ bias, float* __restrict__ out, long total)
{
  long i = (long)blockIdx.x * blockDim.x + threadIdx.x;
  if (i >= total) return;
  int node = (int)(i >> 7), f = (int)(i & 127);
  float di = dinv[node];
  out[i] = h[i] * di * di + bias[f];
}

// out[dst,f] += h[src,f] * dinv[src]*dinv[dst]  (4 floats per thread)
__global__ void gcn_scatter(const float* __restrict__ h, const int* __restrict__ src,
                            const int* __restrict__ dst, const float* __restrict__ dinv,
                            float* __restrict__ out, int E)
{
  long t = (long)blockIdx.x * blockDim.x + threadIdx.x;
  if (t >= (long)E * 32) return;
  int e = (int)(t >> 5), q = (int)(t & 31);
  int s = src[e], d = dst[e];
  float w = dinv[s] * dinv[d];
  float4 hv = ((const float4*)(h + ((long)s << 7)))[q];
  float* op = out + ((long)d << 7) + q * 4;
  atomicAdd(op + 0, hv.x * w);
  atomicAdd(op + 1, hv.y * w);
  atomicAdd(op + 2, hv.z * w);
  atomicAdd(op + 3, hv.w * w);
}

// graph mean-pool of relu(agg) over contiguous 32-node groups -> bf16
__global__ void pool_mean_relu(const float* __restrict__ agg, bf16_t* __restrict__ outbf)
{
  long i = (long)blockIdx.x * blockDim.x + threadIdx.x;
  if (i >= (long)BSZ_ * HID_) return;
  int g = (int)(i >> 7), f = (int)(i & 127);
  float s = 0.0f;
#pragma unroll 4
  for (int j = 0; j < 32; ++j)
    s += fmaxf(agg[(((long)g * 32 + j) << 7) + f], 0.0f);
  outbf[i] = (bf16_t)(s * 0.03125f);
}

// P[e, i*32+k] = bf16(nf[row(e),i] * relu(feat(e)*en1W[k]+en1b[k]));  P[e,4096+i]=nf
__global__ __launch_bounds__(128)
void build_P(const float* __restrict__ nf, const float* __restrict__ hb1,
             const float* __restrict__ hb2, const float* __restrict__ en1W,
             const float* __restrict__ en1b, bf16_t* __restrict__ P)
{
  int e = blockIdx.x;           // 8192 edges
  int t = threadIdx.x;          // 128 features
  int rowi; float feat;
  if      (e < 2048) { rowi = e;        feat = hb1[e]; }
  else if (e < 4096) { rowi = e;        feat = hb1[e - 2048]; }
  else if (e < 6144) { rowi = e - 4096; feat = hb1[e - 4096]; }
  else               { rowi = e - 4096; feat = hb2[e - 6144]; }
  float nfv = nf[(long)rowi * HID_ + t];
  bf16_t* pr = P + (long)e * KPE_;
  pr[4096 + t] = (bf16_t)nfv;
#pragma unroll
  for (int k = 0; k < 32; ++k) {
    float h1 = fmaxf(feat * en1W[k] + en1b[k], 0.0f);
    pr[(long)t * 32 + k] = (bf16_t)(nfv * h1);
  }
}

// nf_conv = relu(msg[cross(c)] + msg[self(c)] + nf@nn_root + nn_bias) -> bf16
__global__ void combine_nfconv(const float* __restrict__ msg, const float* __restrict__ root_out,
                               const float* __restrict__ nn_bias, bf16_t* __restrict__ nfc)
{
  long i = (long)blockIdx.x * blockDim.x + threadIdx.x;
  if (i >= (long)B2_ * HID_) return;
  int c = (int)(i >> 7), o = (int)(i & 127);
  int ecross = (c < 2048) ? (2048 + c) : (c - 2048);
  int eself  = 4096 + c;
  float v = msg[(long)ecross * HID_ + o] + msg[(long)eself * HID_ + o]
          + root_out[i] + nn_bias[o];
  nfc[i] = (bf16_t)fmaxf(v, 0.0f);
}

// PyTorch-order GRU gates (r,z,n); hidden = nf
__global__ void gru_update(const float* __restrict__ gx, const float* __restrict__ gh,
                           const float* __restrict__ nf, float* __restrict__ nf_new)
{
  long i = (long)blockIdx.x * blockDim.x + threadIdx.x;
  if (i >= (long)B2_ * HID_) return;
  int c = (int)(i >> 7), o = (int)(i & 127);
  const float* gxr = gx + (long)c * 384;
  const float* ghr = gh + (long)c * 384;
  float r  = 1.0f / (1.0f + __expf(-(gxr[o]       + ghr[o])));
  float z  = 1.0f / (1.0f + __expf(-(gxr[128 + o] + ghr[128 + o])));
  float nc = tanhf(gxr[256 + o] + r * ghr[256 + o]);
  nf_new[i] = (1.0f - z) * nc + z * nf[i];
}

// xg[b, :] = cat(nf_new[b], nf_new[2048+b]) -> bf16 [2048, 256]
__global__ void build_xg(const float* __restrict__ nf_new, bf16_t* __restrict__ xg)
{
  long i = (long)blockIdx.x * blockDim.x + threadIdx.x;
  if (i >= (long)BSZ_ * 256) return;
  int b = (int)(i >> 8), f = (int)(i & 255);
  float v = (f < 128) ? nf_new[(long)b * 128 + f]
                      : nf_new[(long)(2048 + b) * 128 + (f - 128)];
  xg[i] = (bf16_t)v;
}

__global__ void final_matvec(const float* __restrict__ x, const float* __restrict__ w,
                             const float* __restrict__ b, float* __restrict__ out)
{
  int i = blockIdx.x * blockDim.x + threadIdx.x;
  if (i >= BSZ_) return;
  float s = 0.0f;
#pragma unroll 8
  for (int f = 0; f < 128; ++f) s += x[(long)i * 128 + f] * w[f];
  out[i] = s + b[0];
}

// ---------------------------------------------------------------------------

static inline dim3 gr1(long total, int bs) { return dim3((unsigned)((total + bs - 1) / bs)); }
static inline dim3 gemm_grid(int M, int N) {
  return dim3((unsigned)((M / 64) * (N / 64)));   // 64x64 tile per 4-wave block
}

extern "C" void kernel_launch(void* const* d_in, const int* in_sizes, int n_in,
                              void* d_out, int out_size, void* d_ws, size_t ws_size,
                              hipStream_t stream)
{
  const float* x_side[2]   = { (const float*)d_in[0], (const float*)d_in[1] };
  const int*   ei_side[2]  = { (const int*)d_in[2],  (const int*)d_in[3]  };
  const float* hb1    = (const float*)d_in[6];
  const float* hb2    = (const float*)d_in[7];
  const float* conv1W = (const float*)d_in[9];
  const float* conv1b = (const float*)d_in[10];
  const float* conv2W = (const float*)d_in[11];
  const float* conv2b = (const float*)d_in[12];
  const float* projW  = (const float*)d_in[13];
  const float* projb  = (const float*)d_in[14];
  const float* en1W   = (const float*)d_in[15];
  const float* en1b   = (const float*)d_in[16];
  const float* en2W   = (const float*)d_in[17];
  const float* en2b   = (const float*)d_in[18];
  const float* nnroot = (const float*)d_in[19];
  const float* nnbias = (const float*)d_in[20];
  const float* gihW   = (const float*)d_in[21];
  const float* gihb   = (const float*)d_in[22];
  const float* ghhW   = (const float*)d_in[23];
  const float* ghhb   = (const float*)d_in[24];
  const float* c1W    = (const float*)d_in[25];
  const float* c1b    = (const float*)d_in[26];
  const float* c2W    = (const float*)d_in[27];
  const float* c2b    = (const float*)d_in[28];
  const float* c3W    = (const float*)d_in[29];
  const float* c3b    = (const float*)d_in[30];
  float* out = (float*)d_out;

  // ---- workspace layout (256B aligned bump allocator) ----
  char* ws = (char*)d_ws;
  size_t off = 0;
  auto A_ = [&](size_t bytes) { size_t o = off; off += (bytes + 255) & ~(size_t)255; return o; };
  const size_t o_xpad = A_((size_t)NN_ * KP1_ * 2);   // stage-1 scratch (reused by P)
  const size_t o_hlin = A_((size_t)NN_ * HID_ * 4);
  const size_t o_agg  = A_((size_t)NN_ * HID_ * 4);
  const size_t o_h2   = A_((size_t)NN_ * HID_ * 2);
  const size_t o_deg  = A_((size_t)NN_ * 4);
  const size_t o_nft  = A_((size_t)B2_ * HID_ * 2);   // pooled node features (bf16)
  const size_t o_w1t  = A_((size_t)HID_ * KP1_ * 2);
  const size_t o_w2ct = A_((size_t)HID_ * HID_ * 2);
  const size_t o_prjt = A_((size_t)HID_ * HID_ * 2);
  const size_t o_rtt  = A_((size_t)HID_ * HID_ * 2);
  const size_t o_giht = A_((size_t)384 * HID_ * 2);
  const size_t o_ghht = A_((size_t)384 * HID_ * 2);
  const size_t o_c1t  = A_((size_t)HID_ * 256 * 2);
  const size_t o_c2t  = A_((size_t)HID_ * HID_ * 2);
  const size_t o_w2e  = A_((size_t)HID_ * KPE_ * 2);
  const size_t o_nf   = A_((size_t)B2_ * HID_ * 4);
  const size_t o_nfbf = A_((size_t)B2_ * HID_ * 2);
  const size_t o_msg  = A_((size_t)E2_ * HID_ * 4);
  const size_t o_root = A_((size_t)B2_ * HID_ * 4);
  const size_t o_nfc  = A_((size_t)B2_ * HID_ * 2);
  const size_t o_gx   = A_((size_t)B2_ * 384 * 4);
  const size_t o_gh   = A_((size_t)B2_ * 384 * 4);
  const size_t o_nfn  = A_((size_t)B2_ * HID_ * 4);
  const size_t o_xg   = A_((size_t)BSZ_ * 256 * 2);
  const size_t o_c1o  = A_((size_t)BSZ_ * HID_ * 4);
  const size_t o_c1bf = A_((size_t)BSZ_ * HID_ * 2);
  const size_t o_c2o  = A_((size_t)BSZ_ * HID_ * 4);
  const size_t o_P    = o_xpad;    // 69.2MB P overlays dead 79.7MB stage-1 region
  if (ws_size < off) return;

  bf16_t* xpad = (bf16_t*)(ws + o_xpad);
  float*  hlin = (float*) (ws + o_hlin);
  float*  agg  = (float*) (ws + o_agg);
  bf16_t* h2bf = (bf16_t*)(ws + o_h2);
  float*  deg  = (float*) (ws + o_deg);
  bf16_t* nft  = (bf16_t*)(ws + o_nft);
  bf16_t* w1t  = (bf16_t*)(ws + o_w1t);
  bf16_t* w2ct = (bf16_t*)(ws + o_w2ct);
  bf16_t* prjt = (bf16_t*)(ws + o_prjt);
  bf16_t* rtt  = (bf16_t*)(ws + o_rtt);
  bf16_t* giht = (bf16_t*)(ws + o_giht);
  bf16_t* ghht = (bf16_t*)(ws + o_ghht);
  bf16_t* c1t  = (bf16_t*)(ws + o_c1t);
  bf16_t* c2t  = (bf16_t*)(ws + o_c2t);
  bf16_t* w2e  = (bf16_t*)(ws + o_w2e);
  float*  nf   = (float*) (ws + o_nf);
  bf16_t* nfbf = (bf16_t*)(ws + o_nfbf);
  float*  msg  = (float*) (ws + o_msg);
  float*  root = (float*) (ws + o_root);
  bf16_t* nfc  = (bf16_t*)(ws + o_nfc);
  float*  gx   = (float*) (ws + o_gx);
  float*  gh   = (float*) (ws + o_gh);
  float*  nfn  = (float*) (ws + o_nfn);
  bf16_t* xg   = (bf16_t*)(ws + o_xg);
  float*  c1o  = (float*) (ws + o_c1o);
  bf16_t* c1bf = (bf16_t*)(ws + o_c1bf);
  float*  c2o  = (float*) (ws + o_c2o);
  bf16_t* P    = (bf16_t*)(ws + o_P);

  const int BS = 256;

  // ---- one-shot weight prep (bf16, transposed to [N, Kpad]) ----
  transpose_to_bf16<<<gr1((long)HID_ * KP1_, BS), BS, 0, stream>>>(conv1W, w1t, KIN_, HID_, KP1_);
  transpose_to_bf16<<<gr1((long)HID_ * HID_, BS), BS, 0, stream>>>(conv2W, w2ct, HID_, HID_, HID_);
  transpose_to_bf16<<<gr1((long)HID_ * HID_, BS), BS, 0, stream>>>(projW,  prjt, HID_, HID_, HID_);
  transpose_to_bf16<<<gr1((long)HID_ * HID_, BS), BS, 0, stream>>>(nnroot, rtt,  HID_, HID_, HID_);
  transpose_to_bf16<<<gr1((long)384 * HID_,  BS), BS, 0, stream>>>(gihW,   giht, HID_, 384,  HID_);
  transpose_to_bf16<<<gr1((long)384 * HID_,  BS), BS, 0, stream>>>(ghhW,   ghht, HID_, 384,  HID_);
  transpose_to_bf16<<<gr1((long)HID_ * 256,  BS), BS, 0, stream>>>(c1W,    c1t,  256,  HID_, 256);
  transpose_to_bf16<<<gr1((long)HID_ * HID_, BS), BS, 0, stream>>>(c2W,    c2t,  HID_, HID_, HID_);
  build_w2t<<<gr1((long)HID_ * KPE_, BS), BS, 0, stream>>>(en2W, en2b, w2e);

  // ---- stage 1: two GCN layers + mean pool, per side ----
  for (int s = 0; s < 2; ++s) {
    const float* X   = x_side[s];
    const int*   src = ei_side[s];
    const int*   dst = ei_side[s] + NE_;

    f32_to_bf16_pad<<<gr1((long)NN_ * KP1_, BS), BS, 0, stream>>>(X, xpad, NN_, KIN_, KP1_, 0);
    wmma_gemm_bf16<<<gemm_grid(NN_, HID_), 128, 0, stream>>>(xpad, w1t, nullptr, hlin,
                                                             NN_, HID_, KP1_, 0);
    deg_init<<<gr1(NN_, BS), BS, 0, stream>>>(deg, NN_);
    deg_count<<<gr1(NE_, BS), BS, 0, stream>>>(dst, deg, NE_);
    deg_finalize<<<gr1(NN_, BS), BS, 0, stream>>>(deg, NN_);          // deg -> dinv

    gcn_self<<<gr1((long)NN_ * HID_, BS), BS, 0, stream>>>(hlin, deg, conv1b, agg, (long)NN_ * HID_);
    gcn_scatter<<<gr1((long)NE_ * 32, BS), BS, 0, stream>>>(hlin, src, dst, deg, agg, NE_);

    f32_to_bf16_pad<<<gr1((long)NN_ * HID_, BS), BS, 0, stream>>>(agg, h2bf, NN_, HID_, HID_, 1);
    wmma_gemm_bf16<<<gemm_grid(NN_, HID_), 128, 0, stream>>>(h2bf, w2ct, nullptr, hlin,
                                                             NN_, HID_, HID_, 0);
    gcn_self<<<gr1((long)NN_ * HID_, BS), BS, 0, stream>>>(hlin, deg, conv2b, agg, (long)NN_ * HID_);
    gcn_scatter<<<gr1((long)NE_ * 32, BS), BS, 0, stream>>>(hlin, src, dst, deg, agg, NE_);

    pool_mean_relu<<<gr1((long)BSZ_ * HID_, BS), BS, 0, stream>>>(agg, nft + (size_t)s * BSZ_ * HID_);
  }

  // ---- stage 2: MPNN (projection, fused NNConv GEMM, root, GRU) ----
  wmma_gemm_bf16<<<gemm_grid(B2_, HID_), 128, 0, stream>>>(nft, prjt, projb, nf,
                                                           B2_, HID_, HID_, 1);
  f32_to_bf16_pad<<<gr1((long)B2_ * HID_, BS), BS, 0, stream>>>(nf, nfbf, B2_, HID_, HID_, 0);

  build_P<<<dim3(E2_), 128, 0, stream>>>(nf, hb1, hb2, en1W, en1b, P);
  wmma_gemm_bf16<<<gemm_grid(E2_, HID_), 128, 0, stream>>>(P, w2e, nullptr, msg,
                                                           E2_, HID_, KPE_, 0);
  wmma_gemm_bf16<<<gemm_grid(B2_, HID_), 128, 0, stream>>>(nfbf, rtt, nullptr, root,
                                                           B2_, HID_, HID_, 0);
  combine_nfconv<<<gr1((long)B2_ * HID_, BS), BS, 0, stream>>>(msg, root, nnbias, nfc);

  wmma_gemm_bf16<<<gemm_grid(B2_, 384), 128, 0, stream>>>(nfc,  giht, gihb, gx, B2_, 384, HID_, 0);
  wmma_gemm_bf16<<<gemm_grid(B2_, 384), 128, 0, stream>>>(nfbf, ghht, ghhb, gh, B2_, 384, HID_, 0);
  gru_update<<<gr1((long)B2_ * HID_, BS), BS, 0, stream>>>(gx, gh, nf, nfn);

  // ---- head ----
  build_xg<<<gr1((long)BSZ_ * 256, BS), BS, 0, stream>>>(nfn, xg);
  wmma_gemm_bf16<<<gemm_grid(BSZ_, HID_), 128, 0, stream>>>(xg, c1t, c1b, c1o,
                                                            BSZ_, HID_, 256, 1);
  f32_to_bf16_pad<<<gr1((long)BSZ_ * HID_, BS), BS, 0, stream>>>(c1o, c1bf, BSZ_, HID_, HID_, 0);
  wmma_gemm_bf16<<<gemm_grid(BSZ_, HID_), 128, 0, stream>>>(c1bf, c2t, c2b, c2o,
                                                            BSZ_, HID_, HID_, 1);
  final_matvec<<<gr1(BSZ_, BS), BS, 0, stream>>>(c2o, c3W, c3b, out);
}
